// ComformerConvEqui_18726057411381
// MI455X (gfx1250) — matrix operand order repaired and checked by reading, not compiled, not run on hardware
//
#include <hip/hip_runtime.h>
#include <hip/hip_bf16.h>
#include <math.h>
#include <stdint.h>

typedef __attribute__((ext_vector_type(16))) _Float16 v16h;
typedef __attribute__((ext_vector_type(8)))  _Float16 v8h;
typedef __attribute__((ext_vector_type(8)))  float    v8f;

#define IN_CH  256
#define OUT_CH 256
#define EDIM   128
#define NSC    64
#define NVC    8

// Fused GEMM K dimensions: K = 128*qstride + qstride (bias segment)
#define QS1 64
#define K1  (128*QS1 + QS1)     // 8256
#define K1_STEPS (K1/32)        // 258
#define QS2 96
#define K2  (128*QS2 + QS2)     // 12384
#define K2_STEPS (K2/32)        // 387

__device__ __forceinline__ float splus(float x) {
  return x > 20.0f ? x : log1pf(__expf(x));
}
__device__ __forceinline__ v16h splat16(_Float16 x) {
  v16h v;
#pragma unroll
  for (int i = 0; i < 16; i++) v[i] = x;
  return v;
}
__device__ __forceinline__ v16h join8(v8h lo, v8h hi) {
  return __builtin_shufflevector(lo, hi, 0,1,2,3,4,5,6,7,8,9,10,11,12,13,14,15);
}

// ---------------- elementwise pack kernels ----------------

__global__ void k_cvt16(const float* __restrict__ in, _Float16* __restrict__ out, int n) {
  int i = blockIdx.x * blockDim.x + threadIdx.x;
  if (i < n) out[i] = (_Float16)in[i];
}

// in: (K,N) f32 row-major  ->  out: (N,K) f16 row-major (B^T layout for WMMA)
__global__ void k_transpose16(const float* __restrict__ in, _Float16* __restrict__ out, int K, int N) {
  int i = blockIdx.x * blockDim.x + threadIdx.x;
  if (i < K * N) {
    int n = i / K, k = i % K;
    out[i] = (_Float16)in[k * N + n];
  }
}

// Pack fc1_w2 (128 x 5120) + fc1_b2 (5120) into Bt1 (80 x K1) f16.
// k < 8192: j=k>>6, c=k&63 (c = i index of s). Col n<64 -> W0[c, n]=w2[j, c*64+n];
// n in [64,72) -> W1[c, n-64]=w2[j, 4096+c*8+(n-64)]; n in [72,80) -> W2.
// k >= 8192: bias segment, c=k-8192.
__global__ void k_pack_bt1(const float* __restrict__ w2, const float* __restrict__ b2,
                           _Float16* __restrict__ bt) {
  int i = blockIdx.x * blockDim.x + threadIdx.x;
  if (i >= 80 * K1) return;
  int n = i / K1, k = i % K1;
  float val;
  if (k < 8192) {
    int j = k >> 6, c = k & 63;
    const float* row = w2 + (size_t)j * 5120;
    val = (n < 64) ? row[c * 64 + n]
        : (n < 72) ? row[4096 + c * 8 + (n - 64)]
                   : row[4608 + c * 8 + (n - 72)];
  } else {
    int c = k - 8192;
    val = (n < 64) ? b2[c * 64 + n]
        : (n < 72) ? b2[4096 + c * 8 + (n - 64)]
                   : b2[4608 + c * 8 + (n - 72)];
  }
  bt[i] = (_Float16)val;
}

// Pack fc2_w2 (128 x 5120) + fc2_b2 into Bt2 (64 x K2) f16, q-stride 96 (c<64: V0; 64..71: V1; 72..79: V2; pad 0).
__global__ void k_pack_bt2(const float* __restrict__ w2, const float* __restrict__ b2,
                           _Float16* __restrict__ bt) {
  int i = blockIdx.x * blockDim.x + threadIdx.x;
  if (i >= 64 * K2) return;
  int n = i / K2, k = i % K2;
  float val = 0.0f;
  if (k < 128 * QS2) {
    int j = k / QS2, c = k % QS2;
    const float* row = w2 + (size_t)j * 5120;
    if (c < 64)      val = row[c * 64 + n];
    else if (c < 72) val = row[4096 + (c - 64) * 64 + n];
    else if (c < 80) val = row[4608 + (c - 72) * 64 + n];
  } else {
    int c = k - 128 * QS2;
    if (c < 64)      val = b2[c * 64 + n];
    else if (c < 72) val = b2[4096 + (c - 64) * 64 + n];
    else if (c < 80) val = b2[4608 + (c - 72) * 64 + n];
  }
  bt[i] = (_Float16)val;
}

// ---------------- generic WMMA GEMM ----------------
// C(M,N) = act(A(M,K) @ Bt(N,K)^T + bias) [+ addend]; one wave per 16x64 tile.
// A f16 row-major, Bt f16 row-major (i.e. B transposed). M%16==0, N%64==0, K%32==0.
__global__ __launch_bounds__(32)
void k_gemm(const _Float16* __restrict__ A, const _Float16* __restrict__ Bt,
            const float* __restrict__ bias, const float* __restrict__ addend,
            float* __restrict__ Cf, _Float16* __restrict__ Ch,
            int M, int N, int K, int act) {
  const int lane = threadIdx.x;
  const int m0 = blockIdx.x * 16;
  const int n0 = blockIdx.y * 64;
  const int colL = lane & 15;
  const int aoff = (lane < 16) ? 0 : 16;  // A: K sub-offset per lane half
  const int boff = (lane < 16) ? 0 : 16;  // B: K sub-offset per lane half

  v8f acc[4] = {};
  const _Float16* arow = A + (size_t)(m0 + colL) * K;
  const _Float16* bcol = Bt + (size_t)(n0 + colL) * K;

  for (int kc = 0; kc < K; kc += 32) {
    v8h alo = *(const v8h*)(arow + kc + aoff);
    v8h ahi = *(const v8h*)(arow + kc + aoff + 8);
    v16h av = join8(alo, ahi);
#pragma unroll
    for (int t = 0; t < 4; t++) {
      v16h bv = *(const v16h*)(bcol + (size_t)t * 16 * K + kc + boff);
      acc[t] = __builtin_amdgcn_wmma_f32_16x16x32_f16(false, av, false, bv,
                                                      (short)0, acc[t], false, false);
    }
  }

  const int rbase = (lane < 16) ? 0 : 8;
#pragma unroll
  for (int t = 0; t < 4; t++) {
    int col = n0 + t * 16 + colL;
    float bv = bias ? bias[col] : 0.0f;
#pragma unroll
    for (int v = 0; v < 8; v++) {
      int row = m0 + rbase + v;
      float x = acc[t][v] + bv;
      if (act) x = splus(x);
      if (addend) x += addend[(size_t)row * N + col];
      if (Cf) Cf[(size_t)row * N + col] = x;
      else    Ch[(size_t)row * N + col] = (_Float16)x;
    }
  }
}

// ---------------- fused edge pass 1 ----------------
// Per 16-edge tile: o(16x80) = [P[e,j]*s[e,i] | s[e,i]] @ Bt1, fully in WMMA.
// Cols 0..63: o0 (scaled 1/8). Cols 64..79: u1(8)|u2(8) -> outer with sh1/sh2.
__global__ __launch_bounds__(32)
void k_fused_msg1(const float* __restrict__ h, const _Float16* __restrict__ P,
                  const _Float16* __restrict__ Bt, const int* __restrict__ EI,
                  const float* __restrict__ evec, float* __restrict__ acc1,
                  float* __restrict__ cnt, int E) {
  __shared__ __align__(16) _Float16 q[16][QS1];
  __shared__ __align__(16) _Float16 p[16][128];
  __shared__ float shv[16][8];
  __shared__ int   slds[16];

  const int lane = threadIdx.x;
  const int e0 = blockIdx.x * 16;

  if (lane < 16) {
    int e = e0 + lane;
    slds[lane] = EI[e];
    int dn = EI[E + e];
    const float* hr = h + (size_t)dn * NSC;
#pragma unroll
    for (int i = 0; i < NSC; i++) q[lane][i] = (_Float16)hr[i];
    float vx = evec[e * 3 + 0], vy = evec[e * 3 + 1], vz = evec[e * 3 + 2];
    float inorm = rsqrtf(vx * vx + vy * vy + vz * vz + 1e-12f);
    vx *= inorm; vy *= inorm; vz *= inorm;
    const float s3 = 1.7320508075688772f, s5 = 2.23606797749979f;
    shv[lane][0] = s3 * vx;
    shv[lane][1] = s3 * vy;
    shv[lane][2] = s3 * vz;
    shv[lane][3] = s5 * s3 * vx * vz;
    shv[lane][4] = s5 * s3 * vx * vy;
    shv[lane][5] = s5 * (vy * vy - 0.5f * (vx * vx + vz * vz));
    shv[lane][6] = s5 * s3 * vy * vz;
    shv[lane][7] = s5 * 0.5f * s3 * (vz * vz - vx * vx);
  }
  {
    const uint32_t* src = (const uint32_t*)(P + (size_t)e0 * 128);
    uint32_t* dst = (uint32_t*)&p[0][0];
    for (int i = lane; i < 16 * 128 / 2; i += 32) dst[i] = src[i];
  }
  __syncthreads();

  const int r = lane & 15;
  const int asel = (lane < 16) ? 0 : 1;
  const int boff = (lane < 16) ? 0 : 16;
  const v8h* qr = (const v8h*)&q[r][0];
  v8f acc[5] = {};

  for (int s = 0; s < K1_STEPS; s++) {
    _Float16 pf; int qoff;
    if (s < 256) { pf = p[r][s >> 1]; qoff = (s & 1) * 32; }
    else         { pf = (_Float16)1.0f; qoff = (s - 256) * 32; }
    int qb = qoff >> 3;
    v16h av = join8(qr[qb + 2 * asel], qr[qb + 2 * asel + 1]) * splat16(pf);
    const _Float16* bbase = Bt + (size_t)r * K1 + s * 32 + boff;
#pragma unroll
    for (int t = 0; t < 5; t++) {
      v16h bv = *(const v16h*)(bbase + (size_t)t * 16 * K1);
      acc[t] = __builtin_amdgcn_wmma_f32_16x16x32_f16(false, av, false, bv,
                                                      (short)0, acc[t], false, false);
    }
  }

  const int colL = lane & 15;
  const int rbase = (lane < 16) ? 0 : 8;
  const float c8 = 0.125f;
#pragma unroll
  for (int t = 0; t < 4; t++) {
#pragma unroll
    for (int v = 0; v < 8; v++) {
      int row = rbase + v;
      atomicAdd(&acc1[(size_t)slds[row] * 128 + t * 16 + colL], c8 * acc[t][v]);
    }
  }
#pragma unroll
  for (int v = 0; v < 8; v++) {
    int row = rbase + v;
    int sn = slds[row];
    float u = c8 * acc[4][v];
    if (colL < 8) {
#pragma unroll
      for (int m = 0; m < 3; m++)
        atomicAdd(&acc1[(size_t)sn * 128 + 64 + colL * 3 + m], u * shv[row][m]);
    } else {
      int kk = colL - 8;
#pragma unroll
      for (int m = 0; m < 5; m++)
        atomicAdd(&acc1[(size_t)sn * 128 + 88 + kk * 5 + m], u * shv[row][3 + m]);
    }
  }
  if (lane < 16) atomicAdd(&cnt[slds[lane]], 1.0f);
}

// h1 = scatter_mean + h added to first 64 channels
__global__ void k_h1_final(const float* __restrict__ acc1, const float* __restrict__ cnt,
                           const float* __restrict__ h, float* __restrict__ h1, int n) {
  int i = blockIdx.x * blockDim.x + threadIdx.x;
  if (i >= n * 128) return;
  int r = i >> 7, col = i & 127;
  float v = acc1[i] / fmaxf(cnt[r], 1.0f);
  if (col < 64) v += h[r * 64 + col];
  h1[i] = v;
}

// ---------------- fused edge pass 2 ----------------
__global__ __launch_bounds__(32)
void k_fused_msg2(const float* __restrict__ h1, const _Float16* __restrict__ P,
                  const _Float16* __restrict__ Bt, const int* __restrict__ EI,
                  const float* __restrict__ evec, float* __restrict__ acc2, int E) {
  __shared__ __align__(16) _Float16 q[16][QS2];
  __shared__ __align__(16) _Float16 p[16][128];
  __shared__ int slds[16];

  const int lane = threadIdx.x;
  const int e0 = blockIdx.x * 16;

  if (lane < 16) {
    int e = e0 + lane;
    slds[lane] = EI[e];
    int dn = EI[E + e];
    const float* hr = h1 + (size_t)dn * 128;
    float vx = evec[e * 3 + 0], vy = evec[e * 3 + 1], vz = evec[e * 3 + 2];
    float inorm = rsqrtf(vx * vx + vy * vy + vz * vz + 1e-12f);
    vx *= inorm; vy *= inorm; vz *= inorm;
    const float s3 = 1.7320508075688772f, s5 = 2.23606797749979f;
    float sh1x = s3 * vx, sh1y = s3 * vy, sh1z = s3 * vz;
    float sh2[5];
    sh2[0] = s5 * s3 * vx * vz;
    sh2[1] = s5 * s3 * vx * vy;
    sh2[2] = s5 * (vy * vy - 0.5f * (vx * vx + vz * vz));
    sh2[3] = s5 * s3 * vy * vz;
    sh2[4] = s5 * 0.5f * s3 * (vz * vz - vx * vx);
#pragma unroll
    for (int i = 0; i < 64; i++) q[lane][i] = (_Float16)hr[i];
    const float is3 = 0.5773502691896258f, is5 = 0.4472135954999579f;
#pragma unroll
    for (int i = 0; i < NVC; i++) {
      float a1 = (hr[64 + i * 3 + 0] * sh1x + hr[64 + i * 3 + 1] * sh1y +
                  hr[64 + i * 3 + 2] * sh1z) * is3;
      q[lane][64 + i] = (_Float16)a1;
    }
#pragma unroll
    for (int i = 0; i < NVC; i++) {
      float a2 = 0.0f;
#pragma unroll
      for (int m = 0; m < 5; m++) a2 += hr[88 + i * 5 + m] * sh2[m];
      q[lane][72 + i] = (_Float16)(a2 * is5);
    }
#pragma unroll
    for (int i = 80; i < QS2; i++) q[lane][i] = (_Float16)0.0f;
  }
  {
    const uint32_t* src = (const uint32_t*)(P + (size_t)e0 * 128);
    uint32_t* dst = (uint32_t*)&p[0][0];
    for (int i = lane; i < 16 * 128 / 2; i += 32) dst[i] = src[i];
  }
  __syncthreads();

  const int r = lane & 15;
  const int asel = (lane < 16) ? 0 : 1;
  const int boff = (lane < 16) ? 0 : 16;
  const v8h* qr = (const v8h*)&q[r][0];
  v8f acc[4] = {};

  for (int s = 0; s < K2_STEPS; s++) {
    _Float16 pf; int qoff;
    if (s < 384) { int j = s / 3; pf = p[r][j]; qoff = (s - j * 3) * 32; }
    else         { pf = (_Float16)1.0f; qoff = (s - 384) * 32; }
    int qb = qoff >> 3;
    v16h av = join8(qr[qb + 2 * asel], qr[qb + 2 * asel + 1]) * splat16(pf);
    const _Float16* bbase = Bt + (size_t)r * K2 + s * 32 + boff;
#pragma unroll
    for (int t = 0; t < 4; t++) {
      v16h bv = *(const v16h*)(bbase + (size_t)t * 16 * K2);
      acc[t] = __builtin_amdgcn_wmma_f32_16x16x32_f16(false, av, false, bv,
                                                      (short)0, acc[t], false, false);
    }
  }

  const int colL = lane & 15;
  const int rbase = (lane < 16) ? 0 : 8;
  const float inv80 = 0.11180339887498949f;  // 1/sqrt(80)
#pragma unroll
  for (int t = 0; t < 4; t++) {
#pragma unroll
    for (int v = 0; v < 8; v++) {
      int row = rbase + v;
      atomicAdd(&acc2[(size_t)slds[row] * 64 + t * 16 + colL], inv80 * acc[t][v]);
    }
  }
}

// h2 = scatter_mean; accumulate per-channel sum/sumsq for BN
__global__ void k_h2_stats(const float* __restrict__ acc2, const float* __restrict__ cnt,
                           float* __restrict__ h2, float* __restrict__ ssum,
                           float* __restrict__ ssq, int n) {
  int tid = blockIdx.x * blockDim.x + threadIdx.x;
  int c = tid & 63;
  int stride = (gridDim.x * blockDim.x) >> 6;
  float s = 0.0f, s2 = 0.0f;
  for (int r = tid >> 6; r < n; r += stride) {
    float v = acc2[(size_t)r * 64 + c] / fmaxf(cnt[r], 1.0f);
    h2[(size_t)r * 64 + c] = v;
    s += v; s2 += v * v;
  }
  atomicAdd(&ssum[c], s);
  atomicAdd(&ssq[c], s2);
}

// A2 = sp(batchnorm(h2)) as f16 (input to final GEMM)
__global__ void k_bn_apply(const float* __restrict__ h2, const float* __restrict__ ssum,
                           const float* __restrict__ ssq, const float* __restrict__ gamma,
                           const float* __restrict__ beta, _Float16* __restrict__ a2, int n) {
  int i = blockIdx.x * blockDim.x + threadIdx.x;
  if (i >= n * 64) return;
  int c = i & 63;
  float mu = ssum[c] / (float)n;
  float var = ssq[c] / (float)n - mu * mu;
  float rstd = rsqrtf(var + 1e-5f);
  float x = (h2[i] - mu) * rstd * gamma[c] + beta[c];
  a2[i] = (_Float16)splus(x);
}

// ---------------- host launch ----------------

static inline size_t rup(size_t x) { return (x + 255) & ~(size_t)255; }

extern "C" void kernel_launch(void* const* d_in, const int* in_sizes, int n_in,
                              void* d_out, int out_size, void* d_ws, size_t ws_size,
                              hipStream_t stream) {
  const float* nf     = (const float*)d_in[0];
  const int*   EI     = (const int*)  d_in[1];
  const float* ef     = (const float*)d_in[2];
  const float* evec   = (const float*)d_in[3];
  const float* w_node = (const float*)d_in[4];
  const float* b_node = (const float*)d_in[5];
  const float* w_skip = (const float*)d_in[6];
  const float* b_skip = (const float*)d_in[7];
  const float* f1w1   = (const float*)d_in[8];
  const float* f1b1   = (const float*)d_in[9];
  const float* f1w2   = (const float*)d_in[10];
  const float* f1b2   = (const float*)d_in[11];
  const float* f2w1   = (const float*)d_in[12];
  const float* f2b1   = (const float*)d_in[13];
  const float* f2w2   = (const float*)d_in[14];
  const float* f2b2   = (const float*)d_in[15];
  const float* gamma  = (const float*)d_in[16];
  const float* beta   = (const float*)d_in[17];
  const float* w_out  = (const float*)d_in[18];
  const float* b_out  = (const float*)d_in[19];

  const int Nn = in_sizes[0] / IN_CH;  // 10000 (multiple of 16)
  const int E  = in_sizes[1] / 2;      // 40000 (multiple of 16)

  char* w = (char*)d_ws;
  auto alloc = [&](size_t bytes) { char* p = w; w += rup(bytes); return p; };

  _Float16* nf16  = (_Float16*)alloc((size_t)Nn * IN_CH * 2);
  _Float16* ef16  = (_Float16*)alloc((size_t)E * EDIM * 2);
  _Float16* wnT   = (_Float16*)alloc((size_t)64 * 256 * 2);
  _Float16* wsT   = (_Float16*)alloc((size_t)256 * 256 * 2);
  _Float16* f1w1T = (_Float16*)alloc((size_t)128 * 128 * 2);
  _Float16* f2w1T = (_Float16*)alloc((size_t)128 * 128 * 2);
  _Float16* woT   = (_Float16*)alloc((size_t)256 * 64 * 2);
  _Float16* Bt1   = (_Float16*)alloc((size_t)80 * K1 * 2);
  _Float16* Bt2   = (_Float16*)alloc((size_t)64 * K2 * 2);
  float* h_f      = (float*)alloc((size_t)Nn * 64 * 4);
  float* skip_f   = (float*)alloc((size_t)Nn * 256 * 4);
  _Float16* P1    = (_Float16*)alloc((size_t)E * 128 * 2);
  _Float16* P2    = (_Float16*)alloc((size_t)E * 128 * 2);
  float* acc1     = (float*)alloc((size_t)Nn * 128 * 4);
  float* cnt1     = (float*)alloc((size_t)Nn * 4);
  float* h1       = (float*)alloc((size_t)Nn * 128 * 4);
  float* acc2     = (float*)alloc((size_t)Nn * 64 * 4);
  float* h2       = (float*)alloc((size_t)Nn * 64 * 4);
  _Float16* a2h   = (_Float16*)alloc((size_t)Nn * 64 * 2);
  float* ssum     = (float*)alloc(64 * 4);
  float* ssq      = (float*)alloc(64 * 4);
  (void)ws_size; (void)n_in; (void)out_size;

  hipMemsetAsync(acc1, 0, (size_t)Nn * 128 * 4, stream);
  hipMemsetAsync(cnt1, 0, (size_t)Nn * 4, stream);
  hipMemsetAsync(acc2, 0, (size_t)Nn * 64 * 4, stream);
  hipMemsetAsync(ssum, 0, 64 * 4, stream);
  hipMemsetAsync(ssq,  0, 64 * 4, stream);

  const int TB = 256;
  // conversions + packs
  k_cvt16<<<(Nn * IN_CH + TB - 1) / TB, TB, 0, stream>>>(nf, nf16, Nn * IN_CH);
  k_cvt16<<<(E * EDIM + TB - 1) / TB, TB, 0, stream>>>(ef, ef16, E * EDIM);
  k_transpose16<<<(256 * 64 + TB - 1) / TB, TB, 0, stream>>>(w_node, wnT, 256, 64);
  k_transpose16<<<(256 * 256 + TB - 1) / TB, TB, 0, stream>>>(w_skip, wsT, 256, 256);
  k_transpose16<<<(128 * 128 + TB - 1) / TB, TB, 0, stream>>>(f1w1, f1w1T, 128, 128);
  k_transpose16<<<(128 * 128 + TB - 1) / TB, TB, 0, stream>>>(f2w1, f2w1T, 128, 128);
  k_transpose16<<<(64 * 256 + TB - 1) / TB, TB, 0, stream>>>(w_out, woT, 64, 256);
  k_pack_bt1<<<(80 * K1 + TB - 1) / TB, TB, 0, stream>>>(f1w2, f1b2, Bt1);
  k_pack_bt2<<<(64 * K2 + TB - 1) / TB, TB, 0, stream>>>(f2w2, f2b2, Bt2);

  dim3 blk(32);
  // h = nf @ w_node + b_node ; skip = nf @ w_skip + b_skip
  k_gemm<<<dim3(Nn / 16, 1), blk, 0, stream>>>(nf16, wnT, b_node, nullptr, h_f, nullptr,
                                               Nn, 64, IN_CH, 0);
  k_gemm<<<dim3(Nn / 16, 4), blk, 0, stream>>>(nf16, wsT, b_skip, nullptr, skip_f, nullptr,
                                               Nn, 256, IN_CH, 0);
  // P1 = sp(ef @ fc1_w1 + b1) ; P2 = sp(ef @ fc2_w1 + b1)   (f16 outputs)
  k_gemm<<<dim3(E / 16, 2), blk, 0, stream>>>(ef16, f1w1T, f1b1, nullptr, nullptr, P1,
                                              E, 128, 128, 1);
  k_gemm<<<dim3(E / 16, 2), blk, 0, stream>>>(ef16, f2w1T, f2b1, nullptr, nullptr, P2,
                                              E, 128, 128, 1);
  // message pass 1 (fused generator-GEMM + einsum + scatter)
  k_fused_msg1<<<E / 16, blk, 0, stream>>>(h_f, P1, Bt1, EI, evec, acc1, cnt1, E);
  k_h1_final<<<(Nn * 128 + TB - 1) / TB, TB, 0, stream>>>(acc1, cnt1, h_f, h1, Nn);
  // message pass 2
  k_fused_msg2<<<E / 16, blk, 0, stream>>>(h1, P2, Bt2, EI, evec, acc2, E);
  k_h2_stats<<<40, TB, 0, stream>>>(acc2, cnt1, h2, ssum, ssq, Nn);
  k_bn_apply<<<(Nn * 64 + TB - 1) / TB, TB, 0, stream>>>(h2, ssum, ssq, gamma, beta, a2h, Nn);
  // out = sp(sp(hbn) @ w_out + b_out) + skip
  k_gemm<<<dim3(Nn / 16, 4), blk, 0, stream>>>(a2h, woT, b_out, skip_f, (float*)d_out,
                                               nullptr, Nn, 256, 64, 1);
}
